// mBARTFull_26749056319555
// MI455X (gfx1250) — matrix operand order repaired and checked by reading, not compiled, not run on hardware
//
#include <hip/hip_runtime.h>
#include <math.h>

// ---------------- model constants ----------------
constexpr int cL   = 1024;   // sequence length
constexpr int cN   = 2;      // batch
constexpr int cE   = 1024;   // embed dim
constexpr int cH   = 16;     // heads
constexpr int cDH  = 64;     // head dim
constexpr int cFFN = 4096;
constexpr int cM   = cL * cN;      // 2048 token rows
constexpr int cNE  = cN * cE;      // row stride in (L,N,E) flattening

// ---------------- WMMA types ----------------
typedef __attribute__((ext_vector_type(16))) __bf16 bf16x16;
typedef __attribute__((ext_vector_type(8)))  float  f32x8;

union FragB {
    bf16x16        v;
    uint4          q[2];
    unsigned short s[16];
};

__device__ __forceinline__ f32x8 bwmma(const FragB& a, const FragB& b, f32x8 c) {
    return __builtin_amdgcn_wmma_f32_16x16x32_bf16(false, a.v, false, b.v,
                                                   (short)0, c, false, false);
}

__device__ __forceinline__ unsigned short f2bf(float f) {
    unsigned u = __float_as_uint(f);
    u += 0x7fffu + ((u >> 16) & 1u);          // round-to-nearest-even
    return (unsigned short)(u >> 16);
}

__device__ __forceinline__ uint2 pack4bf(float a, float b, float c, float d) {
    uint2 r;
    r.x = (unsigned)f2bf(a) | ((unsigned)f2bf(b) << 16);
    r.y = (unsigned)f2bf(c) | ((unsigned)f2bf(d) << 16);
    return r;
}

// ---- CDNA5 async global->LDS DMA (tracked by ASYNCcnt), CDNA5 ISA ----
// LDS destination address = low 32 bits of the flat shared-aperture address
// (ISA 10.2).
__device__ __forceinline__ void async_ld128(unsigned lds_off, const void* gaddr) {
    asm volatile("global_load_async_to_lds_b128 %0, %1, off"
                 :: "v"(lds_off), "v"(gaddr)
                 : "memory");
}
__device__ __forceinline__ void wait_async0() {
    asm volatile("s_wait_asynccnt 0x0" ::: "memory");
}

// ---------------- f32 -> bf16 convert (n4 = element count / 4) ----------------
__global__ void cvt_kernel(const float* __restrict__ in,
                           unsigned short* __restrict__ out, long n4) {
    long i = (long)blockIdx.x * blockDim.x + threadIdx.x;
    if (i < n4) {
        float4 v = ((const float4*)in)[i];
        ((uint2*)out)[i] = pack4bf(v.x, v.y, v.z, v.w);
    }
}

// ---------------- LayerNorm: one block per row of E=1024, bf16 out ----------------
__global__ void __launch_bounds__(256)
ln_kernel(const float* __restrict__ x, const float* __restrict__ g,
          const float* __restrict__ b, unsigned short* __restrict__ y) {
    int row = blockIdx.x, tid = threadIdx.x;
    const float* xr = x + (size_t)row * cE;
    float4 v = ((const float4*)xr)[tid];
    float s  = v.x + v.y + v.z + v.w;
    float s2 = v.x * v.x + v.y * v.y + v.z * v.z + v.w * v.w;
    #pragma unroll
    for (int off = 16; off; off >>= 1) {
        s  += __shfl_xor(s,  off, 32);
        s2 += __shfl_xor(s2, off, 32);
    }
    __shared__ float sm[8], sq[8];
    int wid = tid >> 5;
    if ((tid & 31) == 0) { sm[wid] = s; sq[wid] = s2; }
    __syncthreads();
    float S = 0.f, S2 = 0.f;
    #pragma unroll
    for (int i = 0; i < 8; i++) { S += sm[i]; S2 += sq[i]; }
    float mean = S * (1.f / cE);
    float var  = S2 * (1.f / cE) - mean * mean;
    float rstd = rsqrtf(var + 1e-5f);
    float4 gg = ((const float4*)g)[tid];
    float4 bb = ((const float4*)b)[tid];
    uint2 o = pack4bf((v.x - mean) * rstd * gg.x + bb.x,
                      (v.y - mean) * rstd * gg.y + bb.y,
                      (v.z - mean) * rstd * gg.z + bb.z,
                      (v.w - mean) * rstd * gg.w + bb.w);
    ((uint2*)(y + (size_t)row * cE))[tid] = o;
}

// ---------------- GEMM: C[M,Nout] = A[M,K](bf16) @ W[Nout,K]^T (bf16) + bias ----------------
// Tile 128x128 per block (256 threads / 8 waves); wave computes 32x64 via 8 WMMA accs.
// Double-buffered LDS filled by async global->LDS DMA; WMMAs overlap the DMA of
// the next K-tile.
#define LDT 40   // LDS row stride (bf16 elems), pads 32 -> 40 to spread banks

__global__ void __launch_bounds__(256)
gemm_bf16(const unsigned short* __restrict__ A, const unsigned short* __restrict__ W,
          const float* __restrict__ bias, float* __restrict__ resid,
          unsigned short* __restrict__ outb, int K, int Nout, int dogelu) {
    __shared__ __align__(16) unsigned short As[2][128 * LDT];
    __shared__ __align__(16) unsigned short Ws[2][128 * LDT];

    int tid   = threadIdx.x;
    int nbase = blockIdx.x * 128, mbase = blockIdx.y * 128;
    int wave  = tid >> 5, lane = tid & 31;
    int wm    = (wave & 3) * 32, wn = (wave >> 2) * 64;
    int half  = lane >> 4, ln16 = lane & 15;

    const f32x8 zero = {0.f, 0.f, 0.f, 0.f, 0.f, 0.f, 0.f, 0.f};
    f32x8 acc[2][4];
    #pragma unroll
    for (int i = 0; i < 2; i++)
        #pragma unroll
        for (int j = 0; j < 4; j++) acc[i][j] = zero;

    // cooperative fill assignment: each thread DMA-copies 2x16B for A and for W
    int ldrow = tid >> 1;            // 0..127
    int ldch  = (tid & 1) * 16;      // 0 or 16 (k-chunk within 32)
    const unsigned short* pA = A + (size_t)(mbase + ldrow) * K + ldch;
    const unsigned short* pW = W + (size_t)(nbase + ldrow) * K + ldch;
    unsigned asOff[2], wsOff[2];
    #pragma unroll
    for (int b = 0; b < 2; b++) {
        asOff[b] = (unsigned)(size_t)&As[b][ldrow * LDT + ldch];
        wsOff[b] = (unsigned)(size_t)&Ws[b][ldrow * LDT + ldch];
    }

    auto issue_tile = [&](int b, int k0) {
        async_ld128(asOff[b],      pA + k0);
        async_ld128(asOff[b] + 16, pA + k0 + 8);
        async_ld128(wsOff[b],      pW + k0);
        async_ld128(wsOff[b] + 16, pW + k0 + 8);
    };

    int nk = K >> 5;
    int buf = 0;
    issue_tile(0, 0);

    for (int t = 0; t < nk; t++) {
        wait_async0();          // this wave's DMA into As[buf]/Ws[buf] done
        __syncthreads();        // ... and everyone else's

        if (t + 1 < nk) issue_tile(buf ^ 1, (t + 1) * 32);  // overlap with WMMAs

        const unsigned short* Ab = As[buf];
        const unsigned short* Wb = Ws[buf];

        FragB af[2];
        #pragma unroll
        for (int rg = 0; rg < 2; rg++) {
            // A-frag layout (ISA 7.12.2): lanes 0-15 K{0-7,16-23}, lanes 16-31 K{8-15,24-31}
            const unsigned short* p = &Ab[(wm + rg * 16 + ln16) * LDT + half * 8];
            af[rg].q[0] = *(const uint4*)p;
            af[rg].q[1] = *(const uint4*)(p + 16);
        }
        #pragma unroll
        for (int cg = 0; cg < 4; cg++) {
            // B-frag: column n of W^T = row n of W; lanes 0-15 K{0-7,16-23}, lanes 16-31 K{8-15,24-31}
            FragB bf_;
            const unsigned short* p = &Wb[(wn + cg * 16 + ln16) * LDT + half * 8];
            bf_.q[0] = *(const uint4*)p;
            bf_.q[1] = *(const uint4*)(p + 16);
            acc[0][cg] = bwmma(af[0], bf_, acc[0][cg]);
            acc[1][cg] = bwmma(af[1], bf_, acc[1][cg]);
        }
        __syncthreads();        // all reads of buf done before it is re-filled
        buf ^= 1;
    }

    // Epilogue: bias (+ GELU) (+ residual f32) (+ bf16 out)
    #pragma unroll
    for (int rg = 0; rg < 2; rg++)
        #pragma unroll
        for (int cg = 0; cg < 4; cg++) {
            int col = nbase + wn + cg * 16 + ln16;
            float bcol = bias[col];
            #pragma unroll
            for (int r = 0; r < 8; r++) {
                int row = mbase + wm + rg * 16 + r + half * 8;
                float val = acc[rg][cg][r] + bcol;
                if (dogelu) val = 0.5f * val * (1.f + erff(val * 0.70710678118654752f));
                size_t idx = (size_t)row * Nout + col;
                if (resid) resid[idx] += val;
                if (outb)  outb[idx] = f2bf(val);
            }
        }
}

// ---------------- Flash attention: 1 wave = 16 query rows, key blocks of 32 ----------------
// q,k,v,o layouts: (L,N,E) bf16 with head slice at h*DH. grid = (L/64, N*H), block 128.
__global__ void __launch_bounds__(128)
attn_kernel(const unsigned short* __restrict__ q, const unsigned short* __restrict__ k,
            const unsigned short* __restrict__ v, unsigned short* __restrict__ o,
            int causal) {
    __shared__ __align__(16) unsigned short plds[4][16 * 32]; // wave-private P staging
    int wave = threadIdx.x >> 5, lane = threadIdx.x & 31;
    int half = lane >> 4, ln16 = lane & 15;
    int n = blockIdx.y >> 4, h = blockIdx.y & 15;
    int qbase = blockIdx.x * 64 + wave * 16;
    int hoff  = h * cDH;
    unsigned short* pw = plds[wave];

    // Q fragments (2 chunks over DH=64)
    FragB qf[2];
    {
        int qi = qbase + ln16;
        const unsigned short* base = q + (size_t)(qi * cN + n) * cE + hoff;
        #pragma unroll
        for (int dc = 0; dc < 2; dc++) {
            const unsigned short* p = base + dc * 32 + half * 8;
            qf[dc].q[0] = *(const uint4*)p;
            qf[dc].q[1] = *(const uint4*)(p + 16);
        }
    }

    const f32x8 zero = {0.f, 0.f, 0.f, 0.f, 0.f, 0.f, 0.f, 0.f};
    f32x8 Oacc[4] = {zero, zero, zero, zero};
    float mrun[8], lrun[8];
    #pragma unroll
    for (int r = 0; r < 8; r++) { mrun[r] = -1e30f; lrun[r] = 0.f; }

    const float scale = 0.125f; // DH^-0.5
    int kbEnd = causal ? ((qbase + 47) >> 5) : (cL >> 5);

    for (int kb = 0; kb < kbEnd; kb++) {
        // ---- S = (Q K^T) for 16 rows x 32 keys: 4 WMMAs ----
        f32x8 Sa = zero, Sb = zero;
        #pragma unroll
        for (int dc = 0; dc < 2; dc++) {
            FragB k0f, k1f;
            int kj = kb * 32 + ln16;
            const unsigned short* p =
                k + (size_t)(kj * cN + n) * cE + hoff + dc * 32 + half * 8;
            k0f.q[0] = *(const uint4*)p;
            k0f.q[1] = *(const uint4*)(p + 16);
            p += (size_t)16 * cNE; // next 16 keys
            k1f.q[0] = *(const uint4*)p;
            k1f.q[1] = *(const uint4*)(p + 16);
            Sa = bwmma(qf[dc], k0f, Sa);
            Sb = bwmma(qf[dc], k1f, Sb);
        }

        // ---- scale + causal mask ----
        int ca = kb * 32 + ln16;
        #pragma unroll
        for (int r = 0; r < 8; r++) {
            float sa = Sa[r] * scale, sb = Sb[r] * scale;
            if (causal) {
                int rowi = qbase + r + half * 8;
                if (ca > rowi)      sa = -10000.f;
                if (ca + 16 > rowi) sb = -10000.f;
            }
            Sa[r] = sa; Sb[r] = sb;
        }

        // ---- online softmax (reductions within 16-lane halves = one C-frag row group) ----
        #pragma unroll
        for (int r = 0; r < 8; r++) {
            float t = fmaxf(Sa[r], Sb[r]);
            #pragma unroll
            for (int off = 8; off; off >>= 1) t = fmaxf(t, __shfl_xor(t, off, 16));
            float nm   = fmaxf(mrun[r], t);
            float corr = __expf(mrun[r] - nm);
            mrun[r] = nm;
            float pa = __expf(Sa[r] - nm), pb = __expf(Sb[r] - nm);
            Sa[r] = pa; Sb[r] = pb;
            float rs = pa + pb;
            #pragma unroll
            for (int off = 8; off; off >>= 1) rs += __shfl_xor(rs, off, 16);
            lrun[r] = lrun[r] * corr + rs;
            #pragma unroll
            for (int j = 0; j < 4; j++) Oacc[j][r] *= corr;
        }

        // ---- P (16x32 bf16) through wave-private LDS into A-frag layout ----
        #pragma unroll
        for (int r = 0; r < 8; r++) {
            int row = r + half * 8;
            pw[row * 32 + ln16]      = f2bf(Sa[r]);
            pw[row * 32 + 16 + ln16] = f2bf(Sb[r]);
        }
        FragB pf; // DS ops are in-order per wave: loads below see the stores above
        {
            const unsigned short* p = pw + ln16 * 32 + half * 16;
            pf.q[0] = *(const uint4*)p;
            pf.q[1] = *(const uint4*)(p + 8);
        }

        // ---- O += P @ V : 4 WMMAs over DH columns ----
        #pragma unroll
        for (int jg = 0; jg < 4; jg++) {
            FragB vf;
            int d = jg * 16 + ln16;
            const unsigned short* base =
                v + (size_t)((kb * 32 + half * 16) * cN + n) * cE + hoff + d;
            #pragma unroll
            for (int e = 0; e < 16; e++) vf.s[e] = base[(size_t)e * cNE];
            Oacc[jg] = bwmma(pf, vf, Oacc[jg]);
        }
    }

    // ---- normalize & write out ----
    #pragma unroll
    for (int jg = 0; jg < 4; jg++) {
        int d = hoff + jg * 16 + ln16;
        #pragma unroll
        for (int r = 0; r < 8; r++) {
            int rowi = qbase + r + half * 8;
            float val = Oacc[jg][r] / lrun[r];
            o[(size_t)(rowi * cN + n) * cE + d] = f2bf(val);
        }
    }
}

// ---------------- host orchestration ----------------
extern "C" void kernel_launch(void* const* d_in, const int* in_sizes, int n_in,
                              void* d_out, int out_size, void* d_ws, size_t ws_size,
                              hipStream_t stream) {
    (void)in_sizes; (void)n_in; (void)out_size; (void)ws_size;

    const float* src = (const float*)d_in[0];
    const float* tgt = (const float*)d_in[1];

    // enc_params, keys sorted (JAX pytree order): b1,b2,bk,bo,bq,bv,fc1_b,fc1_w,fc2_b,fc2_w,g1,g2,wk,wo,wq,wv
    const float* e_b1  = (const float*)d_in[2];
    const float* e_b2  = (const float*)d_in[3];
    const float* e_bk  = (const float*)d_in[4];
    const float* e_bo  = (const float*)d_in[5];
    const float* e_bq  = (const float*)d_in[6];
    const float* e_bv  = (const float*)d_in[7];
    const float* e_fc1b = (const float*)d_in[8];
    const float* e_fc1w = (const float*)d_in[9];
    const float* e_fc2b = (const float*)d_in[10];
    const float* e_fc2w = (const float*)d_in[11];
    const float* e_g1  = (const float*)d_in[12];
    const float* e_g2  = (const float*)d_in[13];
    const float* e_wk  = (const float*)d_in[14];
    const float* e_wo  = (const float*)d_in[15];
    const float* e_wq  = (const float*)d_in[16];
    const float* e_wv  = (const float*)d_in[17];

    // dec_params sorted: b1,b2,bc,bk,bo,bq,bv,cbk,cbo,cbq,cbv,cwk,cwo,cwq,cwv,
    //                    fc1_b,fc1_w,fc2_b,fc2_w,g1,g2,gc,wk,wo,wq,wv
    const float* dd_b1  = (const float*)d_in[18];
    const float* dd_b2  = (const float*)d_in[19];
    const float* dd_bc  = (const float*)d_in[20];
    const float* dd_bk  = (const float*)d_in[21];
    const float* dd_bo  = (const float*)d_in[22];
    const float* dd_bq  = (const float*)d_in[23];
    const float* dd_bv  = (const float*)d_in[24];
    const float* dd_cbk = (const float*)d_in[25];
    const float* dd_cbo = (const float*)d_in[26];
    const float* dd_cbq = (const float*)d_in[27];
    const float* dd_cbv = (const float*)d_in[28];
    const float* dd_cwk = (const float*)d_in[29];
    const float* dd_cwo = (const float*)d_in[30];
    const float* dd_cwq = (const float*)d_in[31];
    const float* dd_cwv = (const float*)d_in[32];
    const float* dd_fc1b = (const float*)d_in[33];
    const float* dd_fc1w = (const float*)d_in[34];
    const float* dd_fc2b = (const float*)d_in[35];
    const float* dd_fc2w = (const float*)d_in[36];
    const float* dd_g1  = (const float*)d_in[37];
    const float* dd_g2  = (const float*)d_in[38];
    const float* dd_gc  = (const float*)d_in[39];
    const float* dd_wk  = (const float*)d_in[40];
    const float* dd_wo  = (const float*)d_in[41];
    const float* dd_wq  = (const float*)d_in[42];
    const float* dd_wv  = (const float*)d_in[43];

    // workspace bump allocator (everything rewritten each call -> graph-replay safe)
    char* wp = (char*)d_ws;
    auto alloc = [&](size_t bytes) -> void* {
        void* p = (void*)wp;
        wp += (bytes + 255) & ~(size_t)255;
        return p;
    };
    float*          xbuf  = (float*)alloc((size_t)cM * cE * 4);           // residual stream f32
    unsigned short* wbuf  = (unsigned short*)alloc((size_t)cFFN * cE * 2); // bf16 weight staging
    unsigned short* ybf   = (unsigned short*)alloc((size_t)cM * cE * 2);   // LN output bf16
    unsigned short* qbf   = (unsigned short*)alloc((size_t)cM * cE * 2);
    unsigned short* kbf   = (unsigned short*)alloc((size_t)cM * cE * 2);
    unsigned short* vbf   = (unsigned short*)alloc((size_t)cM * cE * 2);
    unsigned short* obf   = (unsigned short*)alloc((size_t)cM * cE * 2);
    unsigned short* hbf   = (unsigned short*)alloc((size_t)cM * cFFN * 2); // FFN hidden bf16
    unsigned short* encbf = (unsigned short*)alloc((size_t)cM * cE * 2);   // enc_out bf16

    auto cvt = [&](const float* in, unsigned short* out, size_t nelem) {
        long n4 = (long)(nelem / 4);
        cvt_kernel<<<dim3((unsigned)((n4 + 255) / 256)), dim3(256), 0, stream>>>(in, out, n4);
    };
    auto gemm = [&](const unsigned short* Abf, const float* Wf, const float* bias,
                    int K, int Nout, float* resid, unsigned short* outb, int dogelu) {
        cvt(Wf, wbuf, (size_t)Nout * K);
        gemm_bf16<<<dim3(Nout / 128, cM / 128), dim3(256), 0, stream>>>(
            Abf, wbuf, bias, resid, outb, K, Nout, dogelu);
    };
    auto ln = [&](const float* g_, const float* b_) {
        ln_kernel<<<dim3(cM), dim3(256), 0, stream>>>(xbuf, g_, b_, ybf);
    };
    auto attn = [&](int causal) {
        attn_kernel<<<dim3(cL / 64, cN * cH), dim3(128), 0, stream>>>(
            qbf, kbf, vbf, obf, causal);
    };

    const size_t oEb = (size_t)cE * cE;       // E*E
    const size_t oFb = (size_t)cFFN * cE;     // FFN*E

    // ================= encoder =================
    hipMemcpyAsync(xbuf, src, (size_t)cM * cE * 4, hipMemcpyDeviceToDevice, stream);
    for (int p = 0; p < 2; p++) {
        size_t oE = (size_t)p * cE, oEE = (size_t)p * oEb;
        size_t oF = (size_t)p * cFFN, oFE = (size_t)p * oFb;
        ln(e_g1 + oE, e_b1 + oE);
        gemm(ybf, e_wq + oEE, e_bq + oE, cE, cE, nullptr, qbf, 0);
        gemm(ybf, e_wk + oEE, e_bk + oE, cE, cE, nullptr, kbf, 0);
        gemm(ybf, e_wv + oEE, e_bv + oE, cE, cE, nullptr, vbf, 0);
        attn(1);
        gemm(obf, e_wo + oEE, e_bo + oE, cE, cE, xbuf, nullptr, 0);
        ln(e_g2 + oE, e_b2 + oE);
        gemm(ybf, e_fc1w + oFE, e_fc1b + oF, cE, cFFN, nullptr, hbf, 1);
        gemm(hbf, e_fc2w + oFE, e_fc2b + oE, cFFN, cE, xbuf, nullptr, 0);
    }
    cvt(xbuf, encbf, (size_t)cM * cE);

    // ================= decoder =================
    hipMemcpyAsync(xbuf, tgt, (size_t)cM * cE * 4, hipMemcpyDeviceToDevice, stream);
    for (int p = 0; p < 2; p++) {
        size_t oE = (size_t)p * cE, oEE = (size_t)p * oEb;
        size_t oF = (size_t)p * cFFN, oFE = (size_t)p * oFb;
        // self-attention (causal)
        ln(dd_g1 + oE, dd_b1 + oE);
        gemm(ybf, dd_wq + oEE, dd_bq + oE, cE, cE, nullptr, qbf, 0);
        gemm(ybf, dd_wk + oEE, dd_bk + oE, cE, cE, nullptr, kbf, 0);
        gemm(ybf, dd_wv + oEE, dd_bv + oE, cE, cE, nullptr, vbf, 0);
        attn(1);
        gemm(obf, dd_wo + oEE, dd_bo + oE, cE, cE, xbuf, nullptr, 0);
        // cross-attention (K/V from encoder output, no mask)
        ln(dd_gc + oE, dd_bc + oE);
        gemm(ybf,   dd_cwq + oEE, dd_cbq + oE, cE, cE, nullptr, qbf, 0);
        gemm(encbf, dd_cwk + oEE, dd_cbk + oE, cE, cE, nullptr, kbf, 0);
        gemm(encbf, dd_cwv + oEE, dd_cbv + oE, cE, cE, nullptr, vbf, 0);
        attn(0);
        gemm(obf, dd_cwo + oEE, dd_cbo + oE, cE, cE, xbuf, nullptr, 0);
        // FFN
        ln(dd_g2 + oE, dd_b2 + oE);
        gemm(ybf, dd_fc1w + oFE, dd_fc1b + oF, cE, cFFN, nullptr, hbf, 1);
        gemm(hbf, dd_fc2w + oFE, dd_fc2b + oE, cFFN, cE, xbuf, nullptr, 0);
    }

    hipMemcpyAsync(d_out, xbuf, (size_t)cM * cE * 4, hipMemcpyDeviceToDevice, stream);
}
